// LocalityEnhancedAttention_12412455485558
// MI455X (gfx1250) — matrix operand order repaired and checked by reading, not compiled, not run on hardware
//
#include <hip/hip_runtime.h>

#define D_MODEL 1024
#define NUM_HEADS 16
#define D_K 64
#define SEQ 2048
#define BATCH 4
#define WIN 16

typedef __attribute__((ext_vector_type(16))) __bf16 v16bf;
typedef __attribute__((ext_vector_type(8)))  __bf16 v8bf;
typedef __attribute__((ext_vector_type(8)))  float  v8f;

__device__ __forceinline__ __bf16 f2bf(float f) {
  unsigned u = __builtin_bit_cast(unsigned, f);
  u += 0x7FFFu + ((u >> 16) & 1u);             // round-to-nearest-even
  unsigned short h = (unsigned short)(u >> 16);
  return __builtin_bit_cast(__bf16, h);
}

__device__ __forceinline__ v16bf mkfrag(v8bf lo, v8bf hi) {
  v16bf r;
#pragma unroll
  for (int i = 0; i < 8; ++i) { r[i] = lo[i]; r[i + 8] = hi[i]; }
  return r;
}

__device__ __forceinline__ v8f wmma_bf16(v16bf a, v16bf b, v8f c) {
  // D = A(16x32 bf16) * B(32x16 bf16) + C(16x16 f32)
  return __builtin_amdgcn_wmma_f32_16x16x32_bf16(false, a, false, b, (short)0, c,
                                                 false, false);
}

// ---- stage 32 contiguous elements into LDS as bf16 ----
__device__ __forceinline__ void load32(const float* __restrict__ g, __bf16* s) {
  const float4* p = (const float4*)g;
#pragma unroll
  for (int i = 0; i < 8; ++i) {
    float4 v = p[i];
    s[4 * i + 0] = f2bf(v.x); s[4 * i + 1] = f2bf(v.y);
    s[4 * i + 2] = f2bf(v.z); s[4 * i + 3] = f2bf(v.w);
  }
}
__device__ __forceinline__ void load32(const __bf16* __restrict__ g, __bf16* s) {
  const v8bf* p = (const v8bf*)g;
  v8bf* d = (v8bf*)s;
#pragma unroll
  for (int i = 0; i < 4; ++i) d[i] = p[i];
}

// =====================================================================
// GEMM: out[M=8192, N=1024] = A[M,1024] @ W[1024,1024]^T + bias
// Block tile 128x128, 8 waves of 64x32, BK=64 (2 WMMA K-steps per stage).
// Next K-tile is prefetched (global_prefetch_b8) while WMMAs run.
// BHSD=true  -> write bf16 in [B*H, S, 64] layout (QKV projections)
// BHSD=false -> write OT (float) row-major [M,N]   (output projection)
// =====================================================================
template <typename AT, typename OT, bool BHSD>
__global__ __launch_bounds__(256) void gemm_wmma(
    const AT* __restrict__ A, const float* __restrict__ W,
    const float* __restrict__ bias, OT* __restrict__ out) {
  constexpr int K = D_MODEL, N = D_MODEL;
  constexpr int LDT = 72;  // 64 + 8 halves pad: 144B row, conflict-free
  __shared__ __align__(16) __bf16 sA[128][LDT];
  __shared__ __align__(16) __bf16 sB[128][LDT];

  const int tid  = threadIdx.x;
  const int lane = tid & 31;
  const int wave = tid >> 5;
  const int lq   = lane & 15;      // row / col within 16
  const int kg   = lane >> 4;      // K-group (half-wave)
  const int wm   = wave >> 2, wn = wave & 3;
  const int mBlk = blockIdx.y * 128, nBlk = blockIdx.x * 128;

  v8f acc[4][2] = {};

  const int ldRow = tid >> 1, ldSeg = (tid & 1) * 32;
  const AT*    aRow = A + (size_t)(mBlk + ldRow) * K + ldSeg;
  const float* wRow = W + (size_t)(nBlk + ldRow) * K + ldSeg;

  for (int kb = 0; kb < K; kb += 64) {
    load32(aRow + kb, &sA[ldRow][ldSeg]);
    load32(wRow + kb, &sB[ldRow][ldSeg]);
    if (kb + 64 < K) {  // pull next stage toward L2/WGP$ while this one computes
      __builtin_prefetch(aRow + kb + 64, 0, 0);
      __builtin_prefetch(wRow + kb + 64, 0, 0);
    }
    __syncthreads();
#pragma unroll
    for (int ks = 0; ks < 2; ++ks) {
      v16bf af[4], bfrg[2];
#pragma unroll
      for (int mi = 0; mi < 4; ++mi) {
        const __bf16* r0 = &sA[wm * 64 + mi * 16 + lq][ks * 32 + kg * 8];
        af[mi] = mkfrag(*(const v8bf*)r0, *(const v8bf*)(r0 + 16));
      }
#pragma unroll
      for (int ni = 0; ni < 2; ++ni) {
        const __bf16* r0 = &sB[wn * 32 + ni * 16 + lq][ks * 32 + kg * 8];
        bfrg[ni] = mkfrag(*(const v8bf*)r0, *(const v8bf*)(r0 + 16));
      }
#pragma unroll
      for (int mi = 0; mi < 4; ++mi)
#pragma unroll
        for (int ni = 0; ni < 2; ++ni)
          acc[mi][ni] = wmma_bf16(af[mi], bfrg[ni], acc[mi][ni]);
    }
    __syncthreads();
  }

#pragma unroll
  for (int mi = 0; mi < 4; ++mi) {
#pragma unroll
    for (int ni = 0; ni < 2; ++ni) {
      const int nglob = nBlk + wn * 32 + ni * 16 + lq;
      const float bb = bias[nglob];
#pragma unroll
      for (int r = 0; r < 8; ++r) {
        const int mglob = mBlk + wm * 64 + mi * 16 + r + kg * 8;
        const float v = acc[mi][ni][r] + bb;
        if constexpr (BHSD) {
          const int b = mglob >> 11, s = mglob & (SEQ - 1);
          const int h = nglob >> 6, d = nglob & 63;
          out[((((size_t)b * NUM_HEADS + h) * SEQ + s) << 6) + d] = f2bf(v);
        } else {
          out[(size_t)mglob * N + nglob] = v;
        }
      }
    }
  }
}

// =====================================================================
// Flash attention: grid (B*H, S/64); 4 waves x 16 query rows each.
// Streams 64-key blocks; online softmax in registers; local banded bias.
// ctx written as bf16 [B, S, H*Dk] (= input layout for final projection).
// =====================================================================
__global__ __launch_bounds__(128) void attn_wmma(
    const __bf16* __restrict__ Q, const __bf16* __restrict__ Kh,
    const __bf16* __restrict__ V, const float* __restrict__ lbias,
    __bf16* __restrict__ ctx) {
  __shared__ __align__(16) __bf16 sVT[64][72];    // [channel][key] (transposed V)
  __shared__ __align__(16) __bf16 sP[4][16][72];  // per-wave P tile [row][key]
  __shared__ float sBias[2 * WIN + 1];

  const int tid  = threadIdx.x;
  const int lane = tid & 31;
  const int wave = tid >> 5;
  const int lq   = lane & 15;
  const int kg   = lane >> 4;
  const int bh   = blockIdx.x;
  const int qbase = blockIdx.y * 64 + wave * 16;
  const size_t headOff = (size_t)bh * SEQ * D_K;

  if (tid < 2 * WIN + 1) sBias[tid] = 2.0f * lbias[tid];  // fold LOCALITY_STRENGTH

  // Q A-fragments (row = lq, K-chunks at kg*8 and 16+kg*8 per ISA layout)
  v16bf qf[2];
#pragma unroll
  for (int ks = 0; ks < 2; ++ks) {
    const __bf16* r0 = Q + headOff + (size_t)(qbase + lq) * D_K + ks * 32 + kg * 8;
    qf[ks] = mkfrag(*(const v8bf*)r0, *(const v8bf*)(r0 + 16));
  }

  float mrow[8], lsum[8];
#pragma unroll
  for (int r = 0; r < 8; ++r) { mrow[r] = -3.0e38f; lsum[r] = 0.f; }
  v8f oacc[4] = {};

  const int vKey = tid >> 1, vC0 = (tid & 1) * 32;

  for (int kb = 0; kb < SEQ; kb += 64) {
    __syncthreads();  // prev iteration's sVT/sP fully consumed; sBias ready
    {   // cooperative transposed stage of the V block
      const v8bf* vp = (const v8bf*)(V + headOff + (size_t)(kb + vKey) * D_K + vC0);
#pragma unroll
      for (int c = 0; c < 4; ++c) {
        v8bf chunk = vp[c];
#pragma unroll
        for (int i = 0; i < 8; ++i) sVT[vC0 + c * 8 + i][vKey] = chunk[i];
      }
    }
    if (kb + 64 < SEQ) {  // look-ahead: next key block of K and V toward cache
      __builtin_prefetch(Kh + headOff + (size_t)(kb + 64 + vKey) * D_K + vC0, 0, 0);
      __builtin_prefetch(V  + headOff + (size_t)(kb + 64 + vKey) * D_K + vC0, 0, 0);
    }
    // ---- scores: S(16x64) = Q @ K^T ; K^T B-frags straight from global ----
    v8f sc[4];
#pragma unroll
    for (int t = 0; t < 4; ++t) {
      const __bf16* kr = Kh + headOff + (size_t)(kb + t * 16 + lq) * D_K;
      v16bf b0 = mkfrag(*(const v8bf*)(kr + kg * 8), *(const v8bf*)(kr + kg * 8 + 16));
      v16bf b1 = mkfrag(*(const v8bf*)(kr + 32 + kg * 8),
                        *(const v8bf*)(kr + 32 + kg * 8 + 16));
      v8f c = {};
      c = wmma_bf16(qf[0], b0, c);
      c = wmma_bf16(qf[1], b1, c);
      sc[t] = c;
    }
    // ---- scale + banded local bias; running row max ----
    float rmax[8];
#pragma unroll
    for (int r = 0; r < 8; ++r) rmax[r] = -3.0e38f;
#pragma unroll
    for (int t = 0; t < 4; ++t) {
      const int j = kb + t * 16 + lq;
#pragma unroll
      for (int r = 0; r < 8; ++r) {
        const int i = qbase + r + kg * 8;
        const int rel = j - i + WIN;
        float s = sc[t][r] * 0.125f;  // 1/sqrt(64)
        if ((unsigned)rel <= 2u * WIN) s += sBias[rel];
        sc[t][r] = s;
        rmax[r] = fmaxf(rmax[r], s);
      }
    }
#pragma unroll
    for (int r = 0; r < 8; ++r)
#pragma unroll
      for (int d = 1; d < 16; d <<= 1)
        rmax[r] = fmaxf(rmax[r], __shfl_xor(rmax[r], d, 32));
    // ---- online softmax update ----
    float rsum[8];
#pragma unroll
    for (int r = 0; r < 8; ++r) {
      const float mNew = fmaxf(mrow[r], rmax[r]);
      const float alpha = __expf(mrow[r] - mNew);
      mrow[r] = mNew;
      lsum[r] *= alpha;
      rsum[r] = 0.f;
#pragma unroll
      for (int t = 0; t < 4; ++t) {
        const float p = __expf(sc[t][r] - mNew);
        sc[t][r] = p;
        rsum[r] += p;
      }
#pragma unroll
      for (int t = 0; t < 4; ++t) oacc[t][r] *= alpha;
    }
#pragma unroll
    for (int r = 0; r < 8; ++r) {
#pragma unroll
      for (int d = 1; d < 16; d <<= 1) rsum[r] += __shfl_xor(rsum[r], d, 32);
      lsum[r] += rsum[r];
    }
    // ---- stash P for C-layout -> A-layout relayout ----
#pragma unroll
    for (int t = 0; t < 4; ++t)
#pragma unroll
      for (int r = 0; r < 8; ++r)
        sP[wave][r + kg * 8][t * 16 + lq] = f2bf(sc[t][r]);
    __syncthreads();  // sVT visible to all waves; orders sP write->read
    // ---- O += P @ V ----
    v16bf pf[2];
#pragma unroll
    for (int ks = 0; ks < 2; ++ks) {
      const __bf16* r0 = &sP[wave][lq][ks * 32 + kg * 8];
      pf[ks] = mkfrag(*(const v8bf*)r0, *(const v8bf*)(r0 + 16));
    }
#pragma unroll
    for (int t = 0; t < 4; ++t)
#pragma unroll
      for (int ks = 0; ks < 2; ++ks) {
        const __bf16* r0 = &sVT[t * 16 + lq][ks * 32 + kg * 8];
        v16bf vf = mkfrag(*(const v8bf*)r0, *(const v8bf*)(r0 + 16));
        oacc[t] = wmma_bf16(pf[ks], vf, oacc[t]);
      }
  }
  // ---- normalize and emit ctx [B, S, H*Dk] bf16 ----
  const int b = bh >> 4, h = bh & 15;
#pragma unroll
  for (int r = 0; r < 8; ++r) {
    const float inv = 1.0f / lsum[r];
    const int s = qbase + r + kg * 8;
    __bf16* dst = ctx + ((size_t)b * SEQ + s) * D_MODEL + h * D_K;
#pragma unroll
    for (int t = 0; t < 4; ++t) dst[t * 16 + lq] = f2bf(oacc[t][r] * inv);
  }
}

extern "C" void kernel_launch(void* const* d_in, const int* in_sizes, int n_in,
                              void* d_out, int out_size, void* d_ws, size_t ws_size,
                              hipStream_t stream) {
  (void)in_sizes; (void)n_in; (void)out_size; (void)ws_size;
  const float* query = (const float*)d_in[0];
  const float* key   = (const float*)d_in[1];
  const float* value = (const float*)d_in[2];
  const float* wq = (const float*)d_in[3];
  const float* bq = (const float*)d_in[4];
  const float* wk = (const float*)d_in[5];
  const float* bk = (const float*)d_in[6];
  const float* wv = (const float*)d_in[7];
  const float* bv = (const float*)d_in[8];
  const float* wo = (const float*)d_in[9];
  const float* bo = (const float*)d_in[10];
  const float* lbias = (const float*)d_in[11];

  char* ws = (char*)d_ws;
  const size_t seg = (size_t)BATCH * NUM_HEADS * SEQ * D_K * sizeof(__bf16); // 16 MiB
  __bf16* Qb = (__bf16*)(ws);
  __bf16* Kb = (__bf16*)(ws + seg);
  __bf16* Vb = (__bf16*)(ws + 2 * seg);
  __bf16* Cx = (__bf16*)(ws + 3 * seg);

  dim3 gg(D_MODEL / 128, (BATCH * SEQ) / 128);  // (8, 64)
  gemm_wmma<float, __bf16, true><<<gg, 256, 0, stream>>>(query, wq, bq, Qb);
  gemm_wmma<float, __bf16, true><<<gg, 256, 0, stream>>>(key,   wk, bk, Kb);
  gemm_wmma<float, __bf16, true><<<gg, 256, 0, stream>>>(value, wv, bv, Vb);

  dim3 ga(BATCH * NUM_HEADS, SEQ / 64);         // (64, 32)
  attn_wmma<<<ga, 128, 0, stream>>>(Qb, Kb, Vb, lbias, Cx);

  gemm_wmma<__bf16, float, false><<<gg, 256, 0, stream>>>(Cx, wo, bo, (float*)d_out);
}
